// _PairwiseMessages_257698038059
// MI455X (gfx1250) — matrix operand order repaired and checked by reading, not compiled, not run on hardware
//
#include <hip/hip_runtime.h>
#include <hip/hip_bf16.h>
#include <hip/hip_fp16.h>

// MI455X / gfx1250: wave32, WMMA f32_16x16x32_f16, 320KB LDS per WGP,
// async global->LDS DMA (ASYNCcnt) for weight staging.
//
// Sizes (fixed by reference): B=4, Nq=Nk=256, D_EQUI=64, D_INV=256,
// D_MSG=128, D_FF=256, D_OUT=128, IN_FEATS=448.

typedef __attribute__((ext_vector_type(16))) _Float16 v16h;
typedef __attribute__((ext_vector_type(8)))  _Float16 v8h;
typedef __attribute__((ext_vector_type(8)))  float    v8f;

#define NB   4
#define NQ   256
#define NK   256
#define DEQ  64
#define DINV 256
#define DMSG 128
#define DFF  256
#define DOUT 128
#define INF_ 448          // IN_FEATS

// LDS strides (in halves), all row strides give 16B-aligned, low-conflict rows
#define FS 456            // feats row stride  (448 + 8 pad)   -> 912 B
#define WS 48             // w1 slice row stride (32 + 16 pad) -> 96 B
#define HS 264            // hidden / w2T row stride (256 + 8) -> 528 B

#define FEATS_H   (128 * FS)            // 58368 halves
#define W1S_H     (256 * WS)            // 12288 halves per buffer
#define HID_H     (128 * HS)            // 33792 halves
#define W2L_H     (128 * HS)            // 33792 halves
#define SMEM_HALVES (FEATS_H + 2 * W1S_H + HID_H + W2L_H)
#define SMEM_BYTES  (SMEM_HALVES * 2)   // 301056 B  (<= 320KB WGP LDS)

#define W1S_BYTEOFF (FEATS_H * 2)       // LDS byte offset of w1 slice buffers
#define W1S_BYTES   (W1S_H * 2)         // 24576 B per buffer

// Async DMA: global -> LDS, 16B per lane, tracked on ASYNCcnt (no VGPR
// round-trip, no loadcnt stall in the compute loop).
#define ASYNC_LDS_B128(ldsoff, voff, sbase, imm)                              \
  asm volatile("global_load_async_to_lds_b128 %0, %1, %2 offset:" #imm        \
               :: "v"(ldsoff), "v"(voff), "s"(sbase) : "memory")
#define WAIT_ASYNC0() asm volatile("s_wait_asynccnt 0" ::: "memory")

static __device__ __forceinline__ v16h cat16(v8h a, v8h b) {
  return __builtin_shufflevector(a, b, 0,1,2,3,4,5,6,7,8,9,10,11,12,13,14,15);
}
static __device__ __forceinline__ v8f splat8(float x) {
  return (v8f){x, x, x, x, x, x, x, x};
}

// ---------------------------------------------------------------------------
// Kernel 1: precompute q_msg/k_msg (f16) and W1^T / W2^T (f16) in workspace.
// ---------------------------------------------------------------------------
__global__ __launch_bounds__(256) void prep_kernel(
    const float* __restrict__ q_inv, const float* __restrict__ k_inv,
    const float* __restrict__ Wq, const float* __restrict__ bq,
    const float* __restrict__ Wk, const float* __restrict__ bk,
    const float* __restrict__ W1, const float* __restrict__ W2,
    _Float16* __restrict__ qmsg, _Float16* __restrict__ kmsg,
    _Float16* __restrict__ w1T,  _Float16* __restrict__ w2T) {
  int idx = blockIdx.x * 256 + threadIdx.x;
  const int NQM = NB * NQ * DMSG;        // 131072
  if (idx < NQM) {
    int bq_i = idx >> 7, col = idx & 127;
    const float* x = q_inv + bq_i * DINV;
    float acc = bq[col];
#pragma unroll 4
    for (int i = 0; i < DINV; ++i) acc = fmaf(x[i], Wq[i * DMSG + col], acc);
    qmsg[idx] = (_Float16)acc;
    return;
  }
  idx -= NQM;
  if (idx < NQM) {
    int bk_i = idx >> 7, col = idx & 127;
    const float* x = k_inv + bk_i * DINV;
    float acc = bk[col];
#pragma unroll 4
    for (int i = 0; i < DINV; ++i) acc = fmaf(x[i], Wk[i * DMSG + col], acc);
    kmsg[idx] = (_Float16)acc;
    return;
  }
  idx -= NQM;
  if (idx < INF_ * DFF) {                // W1 [448][256] -> w1T [256][448]
    int n = idx / INF_, kI = idx - n * INF_;
    w1T[idx] = (_Float16)W1[kI * DFF + n];
    return;
  }
  idx -= INF_ * DFF;
  if (idx < DFF * DOUT) {                // W2 [256][128] -> w2T [128][256]
    int n = idx >> 8, kI = idx & 255;
    w2T[idx] = (_Float16)W2[kI * DOUT + n];
  }
}

// ---------------------------------------------------------------------------
// Kernel 2: fused pair-feature construction + MLP, WMMA f16 -> f32.
// Block = (b, 16 q, 8 k) = 128 pair rows. 8 waves, wave w owns M-tile w.
// ---------------------------------------------------------------------------
__global__ __launch_bounds__(256) void pair_kernel(
    const float* __restrict__ q_equi, const float* __restrict__ k_equi,
    const _Float16* __restrict__ qmsg, const _Float16* __restrict__ kmsg,
    const _Float16* __restrict__ w1T,  const _Float16* __restrict__ w2T,
    const float* __restrict__ b1, const float* __restrict__ b2,
    float* __restrict__ out) {
  extern __shared__ __align__(16) char smem_raw[];
  _Float16* feats = (_Float16*)smem_raw;          // [128][FS]
  _Float16* w1s   = feats + FEATS_H;              // 2 x [256][WS]
  _Float16* hid   = w1s + 2 * W1S_H;              // [128][HS]
  _Float16* w2L   = hid + HID_H;                  // [128][HS]

  const int tid  = threadIdx.x;
  const int lane = tid & 31;
  const int wv   = tid >> 5;                      // wave id = M-tile id
  const int l16  = lane & 15;
  const int hi   = lane >> 4;                     // lane half-group
  const int b  = blockIdx.z;
  const int q0 = blockIdx.y * 16;
  const int k0 = blockIdx.x * 8;

  const unsigned long long w1T_u64 = (unsigned long long)w1T;

  // ---- Phase 0: build feature tile in LDS -------------------------------
  // kick off async DMA of W1 slice kc=0 first so it overlaps feature math
  {
    unsigned dst = W1S_BYTEOFF + (unsigned)tid * (WS * 2);
    unsigned src = (unsigned)tid * (INF_ * 2);
    ASYNC_LDS_B128(dst, src, w1T_u64, 0);
    ASYNC_LDS_B128(dst, src, w1T_u64, 16);
    ASYNC_LDS_B128(dst, src, w1T_u64, 32);
    ASYNC_LDS_B128(dst, src, w1T_u64, 48);
  }
  // 192 per-pair geometric features: 128 pairs x 64 channels
  for (int i = tid; i < 128 * 64; i += 256) {
    int p = i >> 6, d = i & 63;
    int q = q0 + (p >> 3), k = k0 + (p & 7);
    const float* qe = q_equi + ((b * NQ + q) * 3) * DEQ + d;
    const float* ke = k_equi + ((b * NK + k) * 3) * DEQ + d;
    float qx = qe[0], qy = qe[DEQ], qz = qe[2 * DEQ];
    float kx = ke[0], ky = ke[DEQ], kz = ke[2 * DEQ];
    float dot = qx * kx + qy * ky + qz * kz;
    float qq  = qx * qx + qy * qy + qz * qz;
    float kk2 = kx * kx + ky * ky + kz * kz;
    float dist = sqrtf(fmaxf(qq + kk2 - 2.0f * dot, 0.0f));   // |q-k|
    float crs  = sqrtf(fmaxf(qq * kk2 - dot * dot, 0.0f));    // |q x k|
    _Float16* fr = feats + p * FS;
    fr[256 + d] = (_Float16)dot;
    fr[320 + d] = (_Float16)dist;
    fr[384 + d] = (_Float16)crs;
  }
  // q_msg / k_msg broadcast copies (v8h = 16B vectors)
  for (int i = tid; i < 2048; i += 256) {          // 128 rows x 16 v8h
    int p = i >> 4, j = (i & 15) * 8;
    int q = q0 + (p >> 3), k = k0 + (p & 7);
    *(v8h*)(feats + p * FS + j)       = *(const v8h*)(qmsg + (b * NQ + q) * DMSG + j);
    *(v8h*)(feats + p * FS + 128 + j) = *(const v8h*)(kmsg + (b * NK + k) * DMSG + j);
  }
  // W2^T into LDS (read-only B operand for GEMM2)
  for (int i = tid; i < 4096; i += 256) {          // 128 rows x 32 v8h
    int n = i >> 5, j = (i & 31) * 8;
    *(v8h*)(w2L + n * HS + j) = *(const v8h*)(w2T + n * 256 + j);
  }
  WAIT_ASYNC0();
  __syncthreads();

  // ---- GEMM1: [128 x 448] @ [448 x 256], bias-initialized ---------------
  v8f acc[16];
#pragma unroll
  for (int nt = 0; nt < 16; ++nt) acc[nt] = splat8(b1[nt * 16 + l16]);

  // A-fragment source: row M = l16 of this wave's tile; ISA A layout:
  // K = kc*32 + hi*8 + {0..7} and +16..23  -> two 16B ds loads
  const _Float16* arow = feats + (wv * 16 + l16) * FS + hi * 8;

  v16h Acur = cat16(*(const v8h*)(arow), *(const v8h*)(arow + 16));

  for (int kc = 0; kc < 14; ++kc) {
    // async-stage next W1 slice; latency hidden behind the 16 WMMAs below
    if (kc + 1 < 14) {
      unsigned dst = W1S_BYTEOFF + ((unsigned)(kc + 1) & 1) * W1S_BYTES
                   + (unsigned)tid * (WS * 2);
      unsigned src = (unsigned)tid * (INF_ * 2) + (unsigned)(kc + 1) * 64;
      ASYNC_LDS_B128(dst, src, w1T_u64, 0);
      ASYNC_LDS_B128(dst, src, w1T_u64, 16);
      ASYNC_LDS_B128(dst, src, w1T_u64, 32);
      ASYNC_LDS_B128(dst, src, w1T_u64, 48);
    }
    // prefetch next A fragment (feats is static after phase 0)
    v16h Anext = Acur;
    if (kc + 1 < 14) {
      const _Float16* ap = arow + (kc + 1) * 32;
      Anext = cat16(*(const v8h*)(ap), *(const v8h*)(ap + 16));
    }
    // B layout: lane n = l16, K = hi*8 + {0..7} and +16..23 in w1T slice;
    // register double-buffer B so WMMAs don't stall on dscnt==0
    const _Float16* bbase = w1s + (kc & 1) * W1S_H + l16 * WS + hi * 8;
    v16h Bc = cat16(*(const v8h*)(bbase), *(const v8h*)(bbase + 16));
#pragma unroll
    for (int nt = 0; nt < 16; ++nt) {
      v16h Bn = Bc;
      if (nt < 15) {
        const _Float16* bp = bbase + (nt + 1) * 16 * WS;
        Bn = cat16(*(const v8h*)bp, *(const v8h*)(bp + 16));
      }
      acc[nt] = __builtin_amdgcn_wmma_f32_16x16x32_f16(
          false, Acur, false, Bc, (short)0, acc[nt], false, false);
      Bc = Bn;
    }
    Acur = Anext;
    WAIT_ASYNC0();
    __syncthreads();
  }

  // ---- SiLU -> f16 hidden (wave-private LDS rows, no barrier needed) ----
#pragma unroll
  for (int nt = 0; nt < 16; ++nt) {
#pragma unroll
    for (int j = 0; j < 8; ++j) {
      float x = acc[nt][j];
      float h = x / (1.0f + __expf(-x));
      // D layout: lane l holds M = hi*8 + j (VGPR j), N = l16
      hid[(wv * 16 + hi * 8 + j) * HS + nt * 16 + l16] = (_Float16)h;
    }
  }

  // ---- GEMM2: [128 x 256] @ [256 x 128], bias-initialized ---------------
  v8f acc2[8];
#pragma unroll
  for (int nt = 0; nt < 8; ++nt) acc2[nt] = splat8(b2[nt * 16 + l16]);

  const _Float16* arow2 = hid + (wv * 16 + l16) * HS + hi * 8;
  v16h A2 = cat16(*(const v8h*)(arow2), *(const v8h*)(arow2 + 16));
#pragma unroll
  for (int kc = 0; kc < 8; ++kc) {
    v16h A2n = A2;
    if (kc < 7) {
      const _Float16* ap = arow2 + (kc + 1) * 32;
      A2n = cat16(*(const v8h*)(ap), *(const v8h*)(ap + 16));
    }
    const _Float16* bbase2 = w2L + l16 * HS + kc * 32 + hi * 8;
    v16h Bc = cat16(*(const v8h*)(bbase2), *(const v8h*)(bbase2 + 16));
#pragma unroll
    for (int nt = 0; nt < 8; ++nt) {
      v16h Bn = Bc;
      if (nt < 7) {
        const _Float16* bp = bbase2 + (nt + 1) * 16 * HS;
        Bn = cat16(*(const v8h*)bp, *(const v8h*)(bp + 16));
      }
      acc2[nt] = __builtin_amdgcn_wmma_f32_16x16x32_f16(
          false, A2, false, Bc, (short)0, acc2[nt], false, false);
      Bc = Bn;
    }
    A2 = A2n;
  }

  // ---- store f32 output: out[b][q][k][dout] -----------------------------
  float* outb = out + (size_t)((b * NQ + q0) * NK + k0) * DOUT;
#pragma unroll
  for (int nt = 0; nt < 8; ++nt) {
#pragma unroll
    for (int j = 0; j < 8; ++j) {
      int p  = wv * 16 + hi * 8 + j;               // pair index in block
      int ql = p >> 3, kl = p & 7;
      outb[(ql * NK + kl) * DOUT + nt * 16 + l16] = acc2[nt][j];
    }
  }
}

// ---------------------------------------------------------------------------
extern "C" void kernel_launch(void* const* d_in, const int* in_sizes, int n_in,
                              void* d_out, int out_size, void* d_ws, size_t ws_size,
                              hipStream_t stream) {
  const float* q_equi = (const float*)d_in[0];
  const float* q_inv  = (const float*)d_in[1];
  const float* k_equi = (const float*)d_in[2];
  const float* k_inv  = (const float*)d_in[3];
  const float* Wq = (const float*)d_in[4];
  const float* bq = (const float*)d_in[5];
  const float* Wk = (const float*)d_in[6];
  const float* bk = (const float*)d_in[7];
  const float* W1 = (const float*)d_in[8];
  const float* b1 = (const float*)d_in[9];
  const float* W2 = (const float*)d_in[10];
  const float* b2 = (const float*)d_in[11];
  float* out = (float*)d_out;

  // workspace carve-up (f16, 16B-aligned offsets)
  char* ws = (char*)d_ws;
  _Float16* qmsg = (_Float16*)(ws + 0);                        // 4*256*128
  _Float16* kmsg = (_Float16*)(ws + 262144);                   // 4*256*128
  _Float16* w1T  = (_Float16*)(ws + 524288);                   // 256*448
  _Float16* w2T  = (_Float16*)(ws + 753664);                   // 128*256
  // total = 819200 bytes

  // Kernel 1: messages + transposed f16 weights
  {
    const int total = 2 * (NB * NQ * DMSG) + INF_ * DFF + DFF * DOUT; // 409600
    prep_kernel<<<(total + 255) / 256, 256, 0, stream>>>(
        q_inv, k_inv, Wq, bq, Wk, bk, W1, W2, qmsg, kmsg, w1T, w2T);
  }

  // Kernel 2: fused pairwise features + 2-layer MLP via WMMA
  {
    dim3 grid(NK / 8, NQ / 16, NB);   // (32, 16, 4) = 2048 blocks
    pair_kernel<<<grid, 256, SMEM_BYTES, stream>>>(
        q_equi, k_equi, qmsg, kmsg, w1T, w2T, b1, b2, out);
  }
}
